// Block_39513699123558
// MI455X (gfx1250) — matrix operand order repaired and checked by reading, not compiled, not run on hardware
//
#include <hip/hip_runtime.h>

// ---------------------------------------------------------------------------
// ChebConv(K=5) + train-mode BN + ReLU, two layers, on MI455X (gfx1250).
//
// Strategy (see analysis): bf16 storage + V_WMMA_F32_16X16X32_BF16 (fp32
// accumulate), multi-kernel pipeline through L2-resident workspace buffers.
// V=162 padded to VP=192 so the GEMM K-dim is a multiple of 32; zero-padded
// Laplacian rows/cols keep the pad region identically zero through the
// Chebyshev recurrence.
// ---------------------------------------------------------------------------

typedef __attribute__((ext_vector_type(16))) __bf16 v16bf;
typedef __attribute__((ext_vector_type(8)))  float  v8f;

static constexpr int    S_   = 4096;          // X*Y*Z
static constexpr int    V_   = 162;
static constexpr int    VP_  = 192;           // padded V (12 * 16)
static constexpr int    B_   = 2;
static constexpr size_t VS_  = (size_t)V_ * S_;      // 663552
static constexpr float  INV_N_ = 1.0f / (float)((size_t)B_ * V_ * S_); // 1/1327104
static constexpr float  BN_EPS_ = 1e-5f;

// --------------------------- WMMA operand gathers ---------------------------
// A operand (16x32 bf16), contiguous k order used here (not the ISA table order):
//   lanes 0-15: M = lane;      lanes 16-31: M = lane-16 (same rows, upper K half)
//   VGPR j:  K = 2j,2j+1   (+16 for upper lane half)
__device__ inline v16bf load_A_bf16(const __bf16* __restrict__ A, int ldA,
                                    int mBase, int kBase, int lane) {
  const int half = lane >> 4;
  const int m    = mBase + (lane & 15);
  v16bf a;
#pragma unroll
  for (int j = 0; j < 8; ++j) {
    const int k = kBase + 2 * j + (half ? 16 : 0);
    const __bf16* p = A + (size_t)m * ldA + k;
    a[2 * j]     = p[0];
    a[2 * j + 1] = p[1];
  }
  return a;
}

// B operand (32x16 bf16): lanes 0-15 hold cols, K=0..15 packed 2/VGPR;
// lanes 16-31 hold K=16..31 (element e of the vector corresponds to K=koff+e).
__device__ inline v16bf load_B_bf16(const __bf16* __restrict__ Bm, size_t ldB,
                                    int kBase, size_t nBase, int lane) {
  const int    half = lane >> 4;
  const size_t n    = nBase + (size_t)(lane & 15);
  const int    koff = kBase + half * 16;
  v16bf b;
#pragma unroll
  for (int j = 0; j < 8; ++j) {
    b[2 * j]     = Bm[(size_t)(koff + 2 * j)     * ldB + n];
    b[2 * j + 1] = Bm[(size_t)(koff + 2 * j + 1) * ldB + n];
  }
  return b;
}

// ------------------------------ prep kernels --------------------------------
__global__ void prep_lap_kernel(const float* __restrict__ lap, __bf16* __restrict__ lapP) {
  int idx = blockIdx.x * 256 + threadIdx.x;           // over VP*VP
  if (idx >= VP_ * VP_) return;
  int u = idx / VP_, v = idx % VP_;
  float val = (u < V_ && v < V_) ? lap[u * V_ + v] : 0.0f;
  lapP[idx] = (__bf16)val;
}

// w:[K,Fin,Fout] fp32 -> Wt:[32 rows(o)][KFp cols(kf)] bf16, zero padded.
__global__ void prep_w_kernel(const float* __restrict__ w, __bf16* __restrict__ Wt,
                              int F, int KF, int KFp) {
  int idx = blockIdx.x * 256 + threadIdx.x;           // over 32*KFp
  if (idx >= 32 * KFp) return;
  int o = idx / KFp, kf = idx % KFp;
  float val = 0.0f;
  if (kf < KF) {
    int k = kf / F, f = kf % F;
    val = w[(size_t)(k * F + f) * 32 + o];
  }
  Wt[idx] = (__bf16)val;
}

// x:[B,16,162,S] fp32 -> T0:[B,16,VP,S] bf16 (pad rows zero)
__global__ void import_x_kernel(const float* __restrict__ x, __bf16* __restrict__ t0) {
  size_t idx = (size_t)blockIdx.x * 256 + threadIdx.x;   // over B*16*VP*S
  int    s   = (int)(idx & (S_ - 1));
  size_t t   = idx >> 12;
  int    v   = (int)(t % VP_);
  size_t bf  = t / VP_;
  float val = 0.0f;
  if (v < V_) val = x[(bf * V_ + v) * S_ + s];
  t0[idx] = (__bf16)val;
}

__global__ void zero_stats_kernel(float* __restrict__ stats) {
  if (threadIdx.x < 64) stats[threadIdx.x] = 0.0f;
}

// --------------------------- Chebyshev step (WMMA) --------------------------
// tout[bf,u,s] = scale * sum_v lapP[u,v]*tin[bf,v,s]  -  (tprev ? tprev[bf,u,s] : 0)
// One 16x16 fp32 accumulator tile per wave; K loop = VP/32 = 6 bf16 WMMAs.
__global__ __launch_bounds__(128) void cheb_step_kernel(
    const __bf16* __restrict__ tin, const __bf16* __restrict__ tprev,
    __bf16* __restrict__ tout, const __bf16* __restrict__ lapP, float scale) {
  const int lane  = threadIdx.x & 31;
  const int wave  = threadIdx.x >> 5;
  const int sBase = blockIdx.x * 64 + wave * 16;
  const int uBase = blockIdx.y * 16;
  const size_t slab = (size_t)blockIdx.z * VP_ * S_;
  const __bf16* Tin = tin + slab;

  v8f acc = {0.f, 0.f, 0.f, 0.f, 0.f, 0.f, 0.f, 0.f};
#pragma unroll
  for (int kt = 0; kt < VP_ / 32; ++kt) {
    if (kt + 1 < VP_ / 32)  // prefetch next K-panel row (global_prefetch_b8)
      __builtin_prefetch(Tin + (size_t)((kt + 1) * 32 + (lane >> 4) * 16) * S_ + sBase, 0, 1);
    v16bf a = load_A_bf16(lapP, VP_, uBase, kt * 32, lane);
    v16bf b = load_B_bf16(Tin, (size_t)S_, kt * 32, (size_t)sBase, lane);
    acc = __builtin_amdgcn_wmma_f32_16x16x32_bf16(false, a, false, b,
                                                  (short)0, acc, false, false);
  }
  const int half = lane >> 4;
  const int scol = sBase + (lane & 15);
#pragma unroll
  for (int r = 0; r < 8; ++r) {
    const int    u   = uBase + r + 8 * half;      // C/D layout: VGPR r -> M=r(+8)
    const size_t off = slab + (size_t)u * S_ + scol;
    float v = scale * acc[r];
    if (tprev) v -= (float)tprev[off];
    tout[off] = (__bf16)v;
  }
}

// ------------------------------ combine (WMMA) ------------------------------
// Y[b,o,n] = sum_kf Wt[o,kf] * T_{k}[b,f,v,s],  n=(v,s), kf=(k<<Fshift)+f.
__global__ __launch_bounds__(128) void combine_kernel(
    const __bf16* __restrict__ t0, const __bf16* __restrict__ t1,
    const __bf16* __restrict__ t2, const __bf16* __restrict__ t3,
    const __bf16* __restrict__ t4, const __bf16* __restrict__ Wt,
    float* __restrict__ Y, int Fshift, int KF, int KFp) {
  const int    lane  = threadIdx.x & 31;
  const int    wave  = threadIdx.x >> 5;
  const size_t nBase = ((size_t)blockIdx.x * 4 + wave) * 16;
  const int    oBase = blockIdx.y * 16;
  const int    b     = blockIdx.z;
  const int    F     = 1 << Fshift;

  const int    half = lane >> 4;
  const size_t n    = nBase + (size_t)(lane & 15);
  const int    vrow = (int)(n >> 12);          // n = v*S + s, S = 4096
  const int    scol = (int)(n & (S_ - 1));

  v8f acc = {0.f, 0.f, 0.f, 0.f, 0.f, 0.f, 0.f, 0.f};
  for (int kt = 0; kt < KFp / 32; ++kt) {
    v16bf a = load_A_bf16(Wt, KFp, oBase, kt * 32, lane);
    const int koff = kt * 32 + half * 16;
    v16bf bb;
#pragma unroll
    for (int e = 0; e < 16; ++e) {
      const int kf = koff + e;
      __bf16 bv = (__bf16)0.0f;
      if (kf < KF) {
        const int k = kf >> Fshift;
        const int f = kf & (F - 1);
        const __bf16* tk = (k == 0) ? t0 : (k == 1) ? t1 : (k == 2) ? t2
                                                   : (k == 3) ? t3 : t4;
        bv = tk[((size_t)(b * F + f) * VP_ + vrow) * S_ + scol];
      }
      bb[e] = bv;
    }
    acc = __builtin_amdgcn_wmma_f32_16x16x32_bf16(false, a, false, bb,
                                                  (short)0, acc, false, false);
  }
  const int ncol = lane & 15;
#pragma unroll
  for (int r = 0; r < 8; ++r) {
    const int o = oBase + r + 8 * half;
    Y[(size_t)(b * 32 + o) * VS_ + nBase + ncol] = acc[r];
  }
  // NOTE: conv bias omitted — train-mode BN subtracts the per-channel mean,
  // which cancels any constant per-channel bias exactly (and b1/b2 == 0).
}

// ------------------------------- BN stats -----------------------------------
// stats[o] = sum, stats[32+o] = sum of squares, over (b, v<162, s).
__global__ __launch_bounds__(256) void bn_stats_kernel(const float* __restrict__ Y,
                                                       float* __restrict__ stats) {
  const int b = blockIdx.z, o = blockIdx.y, chunk = blockIdx.x;
  const int CH = (int)(VS_ / 32);                     // 20736
  const float* p = Y + (size_t)(b * 32 + o) * VS_ + (size_t)chunk * CH;
  float s = 0.0f, sq = 0.0f;
  for (int i = threadIdx.x; i < CH; i += 256) {
    float v = p[i];
    s += v; sq += v * v;
  }
  __shared__ float sh0[256], sh1[256];
  sh0[threadIdx.x] = s; sh1[threadIdx.x] = sq;
  __syncthreads();
  for (int step = 128; step > 0; step >>= 1) {
    if ((int)threadIdx.x < step) {
      sh0[threadIdx.x] += sh0[threadIdx.x + step];
      sh1[threadIdx.x] += sh1[threadIdx.x + step];
    }
    __syncthreads();
  }
  if (threadIdx.x == 0) {
    atomicAdd(&stats[o],      sh0[0]);
    atomicAdd(&stats[32 + o], sh1[0]);
  }
}

// BN+ReLU -> bf16 padded layout (input of next layer, F=32, pad rows zeroed)
__global__ void bn_apply_mid_kernel(const float* __restrict__ Y,
                                    const float* __restrict__ stats,
                                    const float* __restrict__ g,
                                    const float* __restrict__ be,
                                    __bf16* __restrict__ out) {
  size_t idx = (size_t)blockIdx.x * 256 + threadIdx.x;  // over B*32*VP*S
  const int s = (int)(idx & (S_ - 1));
  size_t t = idx >> 12;
  const int v = (int)(t % VP_);
  size_t bo = t / VP_;
  const int o = (int)(bo % 32);
  const int b = (int)(bo / 32);
  float r = 0.0f;
  if (v < V_) {
    const float y    = Y[(size_t)(b * 32 + o) * VS_ + (size_t)v * S_ + s];
    const float mean = stats[o] * INV_N_;
    const float var  = stats[32 + o] * INV_N_ - mean * mean;
    const float yn   = (y - mean) * rsqrtf(var + BN_EPS_) * g[o] + be[o];
    r = fmaxf(yn, 0.0f);
  }
  out[idx] = (__bf16)r;
}

// BN+ReLU -> final fp32 output; Y and d_out share the [B,32,V,S] flat layout.
__global__ void bn_apply_final_kernel(const float* __restrict__ Y,
                                      const float* __restrict__ stats,
                                      const float* __restrict__ g,
                                      const float* __restrict__ be,
                                      float* __restrict__ out) {
  size_t idx = (size_t)blockIdx.x * 256 + threadIdx.x;  // over B*32*V*S
  size_t t = idx >> 12;
  size_t bo = t / V_;
  const int o = (int)(bo % 32);
  const float y    = Y[idx];
  const float mean = stats[o] * INV_N_;
  const float var  = stats[32 + o] * INV_N_ - mean * mean;
  const float yn   = (y - mean) * rsqrtf(var + BN_EPS_) * g[o] + be[o];
  out[idx] = fmaxf(yn, 0.0f);
}

// ------------------------------- host side ----------------------------------
extern "C" void kernel_launch(void* const* d_in, const int* in_sizes, int n_in,
                              void* d_out, int out_size, void* d_ws, size_t ws_size,
                              hipStream_t stream) {
  (void)in_sizes; (void)n_in; (void)out_size; (void)ws_size;
  const float* x   = (const float*)d_in[0];
  const float* lap = (const float*)d_in[1];
  const float* w1  = (const float*)d_in[2];
  const float* g1  = (const float*)d_in[4];
  const float* be1 = (const float*)d_in[5];
  const float* w2  = (const float*)d_in[6];
  const float* g2  = (const float*)d_in[8];
  const float* be2 = (const float*)d_in[9];

  // workspace carve-up (256B aligned regions)
  char* ws = (char*)d_ws;
  size_t off = 0;
  auto carve = [&](size_t bytes) { char* p = ws + off; off += (bytes + 255) & ~(size_t)255; return p; };
  __bf16* lapP  = (__bf16*)carve((size_t)VP_ * VP_ * 2);
  __bf16* Wt1   = (__bf16*)carve((size_t)32 * 96 * 2);
  __bf16* Wt2   = (__bf16*)carve((size_t)32 * 160 * 2);
  float*  stats = (float*)carve(64 * 4);
  const size_t tbytes = (size_t)B_ * 32 * VP_ * S_ * 2;    // 96 MB each (bf16)
  __bf16* T[5];
  for (int k = 0; k < 5; ++k) T[k] = (__bf16*)carve(tbytes);
  float* Y = (float*)carve((size_t)B_ * 32 * VS_ * 4);     // 162 MB fp32

  // ---- preparation ----
  prep_lap_kernel<<<(VP_ * VP_ + 255) / 256, 256, 0, stream>>>(lap, lapP);
  prep_w_kernel<<<(32 * 96 + 255) / 256, 256, 0, stream>>>(w1, Wt1, 16, 80, 96);
  prep_w_kernel<<<(32 * 160 + 255) / 256, 256, 0, stream>>>(w2, Wt2, 32, 160, 160);
  {
    size_t n = (size_t)B_ * 16 * VP_ * S_;
    import_x_kernel<<<(unsigned)(n / 256), 256, 0, stream>>>(x, T[0]);
  }

  const dim3 blk(128);
  const dim3 cmbGrid((unsigned)(VS_ / 64), 2, B_);          // 10368 x 2 x 2
  const unsigned midBlocks = (unsigned)(((size_t)B_ * 32 * VP_ * S_) / 256);
  const unsigned finBlocks = (unsigned)(((size_t)B_ * 32 * V_ * S_) / 256);

  for (int layer = 0; layer < 2; ++layer) {
    const int F = (layer == 0) ? 16 : 32;
    const dim3 chebGrid(S_ / 64, VP_ / 16, B_ * F);         // 64 x 12 x (32|64)

    // Chebyshev recurrence: T1 = L*T0; Tk = 2*L*T(k-1) - T(k-2)
    cheb_step_kernel<<<chebGrid, blk, 0, stream>>>(T[0], nullptr, T[1], lapP, 1.0f);
    cheb_step_kernel<<<chebGrid, blk, 0, stream>>>(T[1], T[0], T[2], lapP, 2.0f);
    cheb_step_kernel<<<chebGrid, blk, 0, stream>>>(T[2], T[1], T[3], lapP, 2.0f);
    cheb_step_kernel<<<chebGrid, blk, 0, stream>>>(T[3], T[2], T[4], lapP, 2.0f);

    // Combine over (k,f)
    if (layer == 0)
      combine_kernel<<<cmbGrid, blk, 0, stream>>>(T[0], T[1], T[2], T[3], T[4],
                                                  Wt1, Y, /*Fshift=*/4, 80, 96);
    else
      combine_kernel<<<cmbGrid, blk, 0, stream>>>(T[0], T[1], T[2], T[3], T[4],
                                                  Wt2, Y, /*Fshift=*/5, 160, 160);

    // BatchNorm stats + apply
    zero_stats_kernel<<<1, 64, 0, stream>>>(stats);
    bn_stats_kernel<<<dim3(32, 32, B_), 256, 0, stream>>>(Y, stats);
    if (layer == 0)
      bn_apply_mid_kernel<<<midBlocks, 256, 0, stream>>>(Y, stats, g1, be1, T[0]);
    else
      bn_apply_final_kernel<<<finBlocks, 256, 0, stream>>>(Y, stats, g2, be2,
                                                           (float*)d_out);
  }
}
